// Encoder_63797444215413
// MI455X (gfx1250) — matrix operand order repaired and checked by reading, not compiled, not run on hardware
//
#include <hip/hip_runtime.h>
#include <hip/hip_bf16.h>

// ---------------------------------------------------------------------------
// Problem constants (from reference)
// ---------------------------------------------------------------------------
#define HH   1024
#define EE   512
#define BB   128
#define SS   256
#define H3   (3 * HH)       // 3072
#define LN_EPS 1e-5f
#define GEMM_BLOCKS 24      // H3 / 128

#ifndef __has_builtin
#define __has_builtin(x) 0
#endif

#if __has_builtin(__builtin_amdgcn_global_load_async_to_lds_b128)
#define HAVE_ASYNC_LDS 1
#else
#define HAVE_ASYNC_LDS 0
#endif

typedef __attribute__((ext_vector_type(16))) __bf16 v16bf;
typedef __attribute__((ext_vector_type(8)))  float  v8f;
typedef __attribute__((ext_vector_type(4)))  int    v4i_;

// ---------------------------------------------------------------------------
// helpers
// ---------------------------------------------------------------------------
__device__ __forceinline__ unsigned short f2bf(float f) {
    unsigned int u = __float_as_uint(f);
    u += 0x7fffu + ((u >> 16) & 1u);          // round-to-nearest-even
    return (unsigned short)(u >> 16);
}

__device__ __forceinline__ float sigmoidf_(float x) {
    return 1.0f / (1.0f + __expf(-x));
}

// wave32 shuffle reduction + 8-wave combine; red8 = shared float[8]
__device__ __forceinline__ float block_reduce_sum(float v, float* red8) {
#pragma unroll
    for (int off = 16; off > 0; off >>= 1) v += __shfl_xor(v, off, 32);
    const int wave = threadIdx.x >> 5;
    const int lane = threadIdx.x & 31;
    __syncthreads();                      // protect previous use
    if (lane == 0) red8[wave] = v;
    __syncthreads();
    float s = 0.0f;
#pragma unroll
    for (int i = 0; i < 8; ++i) s += red8[i];
    return s;
}

#if HAVE_ASYNC_LDS
__device__ __forceinline__ void async_wait_all() {
#if __has_builtin(__builtin_amdgcn_s_wait_asynccnt)
    __builtin_amdgcn_s_wait_asynccnt(0);
#else
    asm volatile("s_wait_asynccnt 0x0" ::: "memory");
#endif
}
// Address-space rebind via integer cast (low 32 bits of a generic shared
// pointer are the LDS byte offset on AMDGPU). Builtin signature (from clang
// diagnostic): (int4 __device__* /*AS1*/, int4 __shared__* /*AS3*/, Ii, Ii).
__device__ __forceinline__ void async_copy16(const unsigned int* gsrc, unsigned int* ldst) {
    __builtin_amdgcn_global_load_async_to_lds_b128(
        (__attribute__((address_space(1))) v4i_*)(unsigned long long)gsrc,
        (__attribute__((address_space(3))) v4i_*)(unsigned long long)ldst,
        0, 0);
}
#endif

// ---------------------------------------------------------------------------
// Weight pack: f32 (K x N row-major) -> bf16 WMMA-B fragment tiles.
//   word ((nt*KT + kt)*32 + lane)*8 + v  holds bf16 pair (k,k+1) with
//   n = nt*16 + (lane&15), k = kt*32 + (lane>>4)*16 + 2*v
// ---------------------------------------------------------------------------
__global__ __launch_bounds__(256)
void pack_weights_kernel(const float* __restrict__ W, unsigned int* __restrict__ Bt,
                         int K, int N) {
    const int KT = K >> 5;
    const int NT = N >> 4;
    const int total = KT * NT * 32 * 8;
    for (int idx = blockIdx.x * blockDim.x + threadIdx.x; idx < total;
         idx += gridDim.x * blockDim.x) {
        const int v    = idx & 7;
        const int lane = (idx >> 3) & 31;
        const int t    = idx >> 8;
        const int kt   = t % KT;
        const int nt   = t / KT;
        const int n    = (nt << 4) + (lane & 15);
        const int k    = (kt << 5) + ((lane >> 4) << 4) + (v << 1);
        const unsigned short lo = f2bf(W[(size_t)k * N + n]);
        const unsigned short hi = f2bf(W[(size_t)(k + 1) * N + n]);
        Bt[idx] = (unsigned int)lo | ((unsigned int)hi << 16);
    }
}

// ---------------------------------------------------------------------------
// Embedding gather -> bf16 sequence buffer xseq[t][b][e]
// ---------------------------------------------------------------------------
__global__ __launch_bounds__(256)
void gather_emb_kernel(const int* __restrict__ inputs, const float* __restrict__ emb,
                       unsigned short* __restrict__ xseq) {
    const int total = SS * BB * EE;
    for (int idx = blockIdx.x * blockDim.x + threadIdx.x; idx < total;
         idx += gridDim.x * blockDim.x) {
        const int t = idx / (BB * EE);
        const int r = idx % (BB * EE);
        const int b = r / EE;
        const int e = r % EE;
        const int tok = inputs[b * SS + t];
        xseq[idx] = f2bf(emb[(size_t)tok * EE + e]);
    }
}

__global__ __launch_bounds__(256)
void zero_state_kernel(float* __restrict__ h, unsigned short* __restrict__ hA, int n) {
    for (int idx = blockIdx.x * blockDim.x + threadIdx.x; idx < n;
         idx += gridDim.x * blockDim.x) {
        h[idx] = 0.0f;
        hA[idx] = 0;
    }
}

// ---------------------------------------------------------------------------
// Batched WMMA GEMM: up to 4 independent jobs per launch (24 blocks each).
// Per job:  C(128 x 3072, f32) = A(128 x K bf16 row-major) * B(tiled bf16).
// Block = 8 waves; wave owns one 16-col n-tile x all 8 m-tiles.
// A staged via double-buffered LDS chunks (128 x 64 bf16); on gfx1250 the
// staging uses GLOBAL_LOAD_ASYNC_TO_LDS_B128 overlapped with the WMMA chain.
// ---------------------------------------------------------------------------
struct GemmJob { const unsigned short* A; const unsigned int* B; float* C; int K; };
struct GemmBatch { GemmJob j[4]; };

__global__ __launch_bounds__(256)
void wmma_gemm_batch_kernel(GemmBatch batch) {
    __shared__ unsigned int As[2][128 * 32];   // 2 x 16 KB

    const GemmJob jb = batch.j[blockIdx.x / GEMM_BLOCKS];
    const int blk  = blockIdx.x % GEMM_BLOCKS;

    const int tid  = threadIdx.x;
    const int lane = tid & 31;
    const int wave = tid >> 5;
    const int half = lane >> 4;
    const int lm   = lane & 15;
    const int nt   = blk * 8 + wave;          // global 16-col n-tile
    const int K    = jb.K;
    const int KT   = K >> 5;
    const int K2   = K >> 1;
    const unsigned int* A32 = (const unsigned int*)jb.A;

    v8f acc[8];
#pragma unroll
    for (int i = 0; i < 8; ++i)
#pragma unroll
        for (int e = 0; e < 8; ++e) acc[i][e] = 0.0f;

    // ---- stage one 128x64 chunk (4 x b128 per thread) ----
    auto stage = [&](int buf, int kc) {
#pragma unroll
        for (int q = 0; q < 4; ++q) {
            const int g  = tid + (q << 8);        // 0..1023 groups of 4 dwords
            const int m  = g >> 3;
            const int w4 = (g & 7) << 2;
            const unsigned int* src = A32 + (size_t)m * K2 + (kc >> 1) + w4;
            unsigned int* dst = &As[buf][g << 2];
#if HAVE_ASYNC_LDS
            async_copy16(src, dst);
#else
#pragma unroll
            for (int x = 0; x < 4; ++x) dst[x] = src[x];
#endif
        }
    };

    stage(0, 0);
#if HAVE_ASYNC_LDS
    async_wait_all();
#endif
    __syncthreads();

    int buf = 0;
    for (int kc = 0; kc < K; kc += 64) {
        if (kc + 64 < K) stage(buf ^ 1, kc + 64);   // prefetch next chunk

#pragma unroll
        for (int kl = 0; kl < 2; ++kl) {
            union U { v16bf v; unsigned int u[8]; };
            U bf;
            const unsigned int* bp =
                jb.B + ((size_t)(nt * KT + (kc >> 5) + kl) * 32 + lane) * 8;
#pragma unroll
            for (int v = 0; v < 8; ++v) bf.u[v] = bp[v];

            U af[8];
#pragma unroll
            for (int mt = 0; mt < 8; ++mt) {
                const int m = (mt << 4) + lm;
#pragma unroll
                for (int v = 0; v < 8; ++v) {
                    const int idx = v + (half << 3) + (kl << 4);
                    af[mt].u[v] = As[buf][(m << 5) + idx];
                }
            }
#pragma unroll
            for (int mt = 0; mt < 8; ++mt)
                acc[mt] = __builtin_amdgcn_wmma_f32_16x16x32_bf16(
                    false, af[mt].v, false, bf.v, (short)0, acc[mt], false, false);
        }

#if HAVE_ASYNC_LDS
        async_wait_all();
#endif
        __syncthreads();
        buf ^= 1;
    }

    // store: C/D layout -> lane<16: M=r, N=lane ; lane>=16: M=r+8, N=lane-16
    const int n = (nt << 4) + lm;
#pragma unroll
    for (int mt = 0; mt < 8; ++mt)
#pragma unroll
        for (int r = 0; r < 8; ++r) {
            const int m = (mt << 4) + r + (half << 3);
            jb.C[(size_t)m * H3 + n] = acc[mt][r];
        }
}

// ---------------------------------------------------------------------------
// Fused per-gate LayerNorm + GRU gates (+ optional mid-LayerNorm), two
// directions per launch (blockIdx < 128 -> job0, else job1).
// ---------------------------------------------------------------------------
struct GateJob {
    const float* ax_pre;   // null => shared mode (ax = axc)
    const float* axc;      // shared-cell bx (3H)
    const float* gx; const float* bx;
    const float* ah_pre;
    const float* gh; const float* bh;
    float* hState;
    unsigned short* hA;
    const float* midg; const float* midb;
    int do_midln;
    float* ctx; int ctxStride;
};
struct GateBatch { GateJob g[2]; };

__device__ __forceinline__ void gate_body(const GateJob& jb, int b, float* red8) {
    const int tid = threadIdx.x;
    const bool cell0 = (jb.ax_pre != nullptr);
    const float invH = 1.0f / (float)HH;

    float axv[3][4], ahv[3][4];
    float aS[3] = {0, 0, 0}, aQ[3] = {0, 0, 0};
    float hS[3] = {0, 0, 0}, hQ[3] = {0, 0, 0};

#pragma unroll
    for (int c = 0; c < 3; ++c)
#pragma unroll
        for (int q = 0; q < 4; ++q) {
            const int j = tid + (q << 8);
            const float hv = jb.ah_pre[(size_t)b * H3 + c * HH + j];
            ahv[c][q] = hv; hS[c] += hv; hQ[c] += hv * hv;
            if (cell0) {
                const float av = jb.ax_pre[(size_t)b * H3 + c * HH + j];
                axv[c][q] = av; aS[c] += av; aQ[c] += av * av;
            }
        }

    float hmu[3], hrs[3], amu[3], ars[3];
#pragma unroll
    for (int c = 0; c < 3; ++c) {
        const float s  = block_reduce_sum(hS[c], red8);
        const float sq = block_reduce_sum(hQ[c], red8);
        hmu[c] = s * invH;
        hrs[c] = __frsqrt_rn(sq * invH - hmu[c] * hmu[c] + LN_EPS);
    }
    if (cell0) {
#pragma unroll
        for (int c = 0; c < 3; ++c) {
            const float s  = block_reduce_sum(aS[c], red8);
            const float sq = block_reduce_sum(aQ[c], red8);
            amu[c] = s * invH;
            ars[c] = __frsqrt_rn(sq * invH - amu[c] * amu[c] + LN_EPS);
        }
    }

    float hnew[4];
    float mS = 0.0f, mQ = 0.0f;
#pragma unroll
    for (int q = 0; q < 4; ++q) {
        const int j = tid + (q << 8);
        float xg[3], hg[3];
#pragma unroll
        for (int c = 0; c < 3; ++c) {
            const int col = c * HH + j;
            xg[c] = cell0 ? ((axv[c][q] - amu[c]) * ars[c] * jb.gx[col] + jb.bx[col])
                          : jb.axc[col];
            hg[c] = (ahv[c][q] - hmu[c]) * hrs[c] * jb.gh[col] + jb.bh[col];
        }
        const float r = sigmoidf_(xg[0] + hg[0]);
        const float z = sigmoidf_(xg[1] + hg[1]);
        const float n = tanhf(xg[2] + r * hg[2]);
        const float hin = jb.hState[(size_t)b * HH + j];
        hnew[q] = (1.0f - z) * n + z * hin;
        mS += hnew[q]; mQ += hnew[q] * hnew[q];
    }

    if (jb.do_midln) {
        const float s  = block_reduce_sum(mS, red8);
        const float sq = block_reduce_sum(mQ, red8);
        const float mu = s * invH;
        const float rs = __frsqrt_rn(sq * invH - mu * mu + LN_EPS);
#pragma unroll
        for (int q = 0; q < 4; ++q) {
            const int j = tid + (q << 8);
            const float hl = (hnew[q] - mu) * rs * jb.midg[j] + jb.midb[j];
            jb.hState[(size_t)b * HH + j] = hl;
            jb.hA[(size_t)b * HH + j]     = f2bf(hl);
        }
    } else {
#pragma unroll
        for (int q = 0; q < 4; ++q) {
            const int j = tid + (q << 8);
            jb.hState[(size_t)b * HH + j] = hnew[q];
            jb.hA[(size_t)b * HH + j]     = f2bf(hnew[q]);
            if (jb.ctx) jb.ctx[(size_t)b * jb.ctxStride + j] = hnew[q];
        }
    }
}

__global__ __launch_bounds__(256)
void gru_gate_batch_kernel(GateBatch batch) {
    __shared__ float red8[8];
    const int which = blockIdx.x >> 7;        // 0..1
    const int b     = blockIdx.x & 127;
    gate_body(batch.g[which], b, red8);
}

// ---------------------------------------------------------------------------
// host side
// ---------------------------------------------------------------------------
extern "C" void kernel_launch(void* const* d_in, const int* in_sizes, int n_in,
                              void* d_out, int out_size, void* d_ws, size_t ws_size,
                              hipStream_t stream) {
    (void)in_sizes; (void)n_in; (void)out_size; (void)ws_size;

    const int*   inputs = (const int*)d_in[0];
    const float* emb    = (const float*)d_in[1];
    const float* f0_Wx = (const float*)d_in[2];
    const float* f0_Wh = (const float*)d_in[3];
    const float* f0_gx = (const float*)d_in[4];
    const float* f0_bx = (const float*)d_in[5];
    const float* f0_gh = (const float*)d_in[6];
    const float* f0_bh = (const float*)d_in[7];
    const float* fs_Wh = (const float*)d_in[9];
    const float* fs_bx = (const float*)d_in[11];
    const float* fs_gh = (const float*)d_in[12];
    const float* fs_bh = (const float*)d_in[13];
    const float* f_lng = (const float*)d_in[14];
    const float* f_lnb = (const float*)d_in[15];
    const float* r0_Wx = (const float*)d_in[16];
    const float* r0_Wh = (const float*)d_in[17];
    const float* r0_gx = (const float*)d_in[18];
    const float* r0_bx = (const float*)d_in[19];
    const float* r0_gh = (const float*)d_in[20];
    const float* r0_bh = (const float*)d_in[21];
    const float* rs_Wh = (const float*)d_in[23];
    const float* rs_bx = (const float*)d_in[25];
    const float* rs_gh = (const float*)d_in[26];
    const float* rs_bh = (const float*)d_in[27];
    const float* r_lng = (const float*)d_in[28];
    const float* r_lnb = (const float*)d_in[29];

    float* out = (float*)d_out;

    // ---- workspace carve-up ----
    char* ws = (char*)d_ws;
    size_t off = 0;
    auto take = [&](size_t bytes) { char* p = ws + off; off += (bytes + 255) & ~(size_t)255; return p; };

    unsigned short* xseq = (unsigned short*)take((size_t)SS * BB * EE * 2);   // 32 MB
    unsigned int* wxf  = (unsigned int*)take((size_t)EE * H3 * 2);
    unsigned int* wh0f = (unsigned int*)take((size_t)HH * H3 * 2);
    unsigned int* whsf = (unsigned int*)take((size_t)HH * H3 * 2);
    unsigned int* wxr  = (unsigned int*)take((size_t)EE * H3 * 2);
    unsigned int* wh0r = (unsigned int*)take((size_t)HH * H3 * 2);
    unsigned int* whsr = (unsigned int*)take((size_t)HH * H3 * 2);
    float* axf = (float*)take((size_t)BB * H3 * 4);
    float* ahf = (float*)take((size_t)BB * H3 * 4);
    float* axr = (float*)take((size_t)BB * H3 * 4);
    float* ahr = (float*)take((size_t)BB * H3 * 4);
    float* hF  = (float*)take((size_t)BB * HH * 4);
    float* hR  = (float*)take((size_t)BB * HH * 4);
    unsigned short* hAF = (unsigned short*)take((size_t)BB * HH * 2);
    unsigned short* hAR = (unsigned short*)take((size_t)BB * HH * 2);

    // ---- one-time prep ----
    pack_weights_kernel<<<1024, 256, 0, stream>>>(f0_Wx, wxf,  EE, H3);
    pack_weights_kernel<<<2048, 256, 0, stream>>>(f0_Wh, wh0f, HH, H3);
    pack_weights_kernel<<<2048, 256, 0, stream>>>(fs_Wh, whsf, HH, H3);
    pack_weights_kernel<<<1024, 256, 0, stream>>>(r0_Wx, wxr,  EE, H3);
    pack_weights_kernel<<<2048, 256, 0, stream>>>(r0_Wh, wh0r, HH, H3);
    pack_weights_kernel<<<2048, 256, 0, stream>>>(rs_Wh, whsr, HH, H3);
    gather_emb_kernel<<<4096, 256, 0, stream>>>(inputs, emb, xseq);
    zero_state_kernel<<<256, 256, 0, stream>>>(hF, hAF, BB * HH);
    zero_state_kernel<<<256, 256, 0, stream>>>(hR, hAR, BB * HH);

    // ---- recurrence: 6 launches per timestep ----
    for (int t = 0; t < SS; ++t) {
        const unsigned short* xf = xseq + (size_t)t * BB * EE;
        const unsigned short* xr = xseq + (size_t)(SS - 1 - t) * BB * EE;
        float* ctxF = out + (size_t)t * BB * 2 * HH;
        float* ctxR = out + (size_t)(SS - 1 - t) * BB * 2 * HH + HH;

        // cell0: x-GEMM + h-GEMM for both directions in one launch
        GemmBatch g4;
        g4.j[0] = { xf,  wxf,  axf, EE };
        g4.j[1] = { hAF, wh0f, ahf, HH };
        g4.j[2] = { xr,  wxr,  axr, EE };
        g4.j[3] = { hAR, wh0r, ahr, HH };
        wmma_gemm_batch_kernel<<<4 * GEMM_BLOCKS, 256, 0, stream>>>(g4);

        GateBatch gb0;
        gb0.g[0] = { axf, nullptr, f0_gx, f0_bx, ahf, f0_gh, f0_bh,
                     hF, hAF, f_lng, f_lnb, 1, nullptr, 0 };
        gb0.g[1] = { axr, nullptr, r0_gx, r0_bx, ahr, r0_gh, r0_bh,
                     hR, hAR, r_lng, r_lnb, 1, nullptr, 0 };
        gru_gate_batch_kernel<<<2 * BB, 256, 0, stream>>>(gb0);

        for (int l = 0; l < 2; ++l) {
            const int last = (l == 1);
            GemmBatch g2;
            g2.j[0] = { hAF, whsf, ahf, HH };
            g2.j[1] = { hAR, whsr, ahr, HH };
            g2.j[2] = g2.j[0]; g2.j[3] = g2.j[1];   // unused slots
            wmma_gemm_batch_kernel<<<2 * GEMM_BLOCKS, 256, 0, stream>>>(g2);

            GateBatch gbs;
            gbs.g[0] = { nullptr, fs_bx, nullptr, nullptr, ahf, fs_gh, fs_bh,
                         hF, hAF, f_lng, f_lnb, last ? 0 : 1,
                         last ? ctxF : nullptr, 2 * HH };
            gbs.g[1] = { nullptr, rs_bx, nullptr, nullptr, ahr, rs_gh, rs_bh,
                         hR, hAR, r_lng, r_lnb, last ? 0 : 1,
                         last ? ctxR : nullptr, 2 * HH };
            gru_gate_batch_kernel<<<2 * BB, 256, 0, stream>>>(gbs);
        }
    }

    // final forward hidden -> d_out tail
    (void)hipMemcpyAsync(out + (size_t)SS * BB * 2 * HH, hF,
                         (size_t)BB * HH * sizeof(float),
                         hipMemcpyDeviceToDevice, stream);
}
